// Sequence_90864328114902
// MI455X (gfx1250) — compile-verified, matrix operand order checked by reading, not run on hardware
//
#include <hip/hip_runtime.h>

// Problem sizes fixed by the reference
#define B_    128   // batch
#define T_    256   // timesteps
#define H_    1024  // hidden
#define NBLK  64    // persistent workgroups: each owns 16 hidden columns
#define NTHR  128   // 4 waves; wave w owns batch rows [32w, 32w+32) = 2 row tiles
#define KP    128   // K-panel staged in LDS per pipeline stage
#define NP    (H_ / KP)        // 8 panels per GEMM
#define PANEL_USH (4 * 16 * KP)        // 8192 ushorts = 16KB per panel (4 gate tiles)

typedef __attribute__((ext_vector_type(16))) __bf16 v16bf;
typedef __attribute__((ext_vector_type(8)))  __bf16 v8bf;
typedef __attribute__((ext_vector_type(8)))  float  v8f;

// Types matching the async-LDS builtin signature: int __vector(4) in AS1/AS3
typedef int i32x4 __attribute__((__vector_size__(4 * sizeof(int))));
typedef __attribute__((address_space(1))) i32x4 gbl_i32x4;
typedef __attribute__((address_space(3))) i32x4 lds_i32x4;

#if defined(__has_builtin)
#if __has_builtin(__builtin_amdgcn_global_load_async_to_lds_b128) && \
    __has_builtin(__builtin_amdgcn_s_wait_asynccnt)
#define USE_ASYNC_LDS 1
#endif
#endif
#ifndef USE_ASYNC_LDS
#define USE_ASYNC_LDS 0
#endif

#define WMMA_BF16(A, Bv, C) \
  __builtin_amdgcn_wmma_f32_16x16x32_bf16(false, (A), false, (Bv), (short)0, (C), false, false)

__device__ __forceinline__ unsigned short f2bf(float f) {
  unsigned u = __float_as_uint(f);
  u += 0x7FFFu + ((u >> 16) & 1u);      // round-to-nearest-even
  return (unsigned short)(u >> 16);
}

// A-fragment: per lane two contiguous 16B chunks at p and p+16 (elements)
__device__ __forceinline__ v16bf load_afrag(const unsigned short* p) {
  union { v16bf v; v8bf h[2]; } u;
  u.h[0] = *reinterpret_cast<const v8bf*>(p);
  u.h[1] = *reinterpret_cast<const v8bf*>(p + 16);
  return u.v;
}

// B-fragment from LDS: frag block = 512 ushorts; lane l owns [l*16, l*16+16)
__device__ __forceinline__ v16bf read_bfrag(const unsigned short* Bs, int buf, int kc, int g, int lane) {
  const unsigned short* p = Bs + buf * PANEL_USH + (((kc << 2) | g) << 9) + lane * 16;
  union { v16bf v; v8bf h[2]; } u;
  u.h[0] = *reinterpret_cast<const v8bf*>(p);
  u.h[1] = *reinterpret_cast<const v8bf*>(p + 8);
  return u.v;
}

__device__ __forceinline__ float sigm(float x)  { return 1.0f / (1.0f + __expf(-x)); }
__device__ __forceinline__ float tanh_(float x) { return 2.0f / (1.0f + __expf(-2.0f * x)) - 1.0f; }

// Device-wide barrier: monotonically increasing counter, goal = NBLK * phase
__device__ __forceinline__ void grid_sync(unsigned* cnt, unsigned goal) {
  __threadfence();
  __syncthreads();
  if (threadIdx.x == 0) {
    atomicAdd(cnt, 1u);
    while (*(volatile unsigned*)cnt < goal) __builtin_amdgcn_s_sleep(1);
    __threadfence();
  }
  __syncthreads();
}

__device__ __forceinline__ void wait_async_panel(void) {
#if USE_ASYNC_LDS
  __builtin_amdgcn_s_wait_asynccnt(8);   // our 8 in-flight ops are the *next* panel's
#endif
}
__device__ __forceinline__ void wait_async_all(void) {
#if USE_ASYNC_LDS
  __builtin_amdgcn_s_wait_asynccnt(0);
#endif
}

// Stage one K-panel of B (4 gate tiles x 16 cols x KP) into LDS in fragment layout.
// 128 threads x 8 transfers x 16B = 16KB. Async (ASYNCcnt path) when available.
__device__ __forceinline__ void issue_panel(const unsigned short* __restrict__ wbf,
                                            unsigned short* __restrict__ Bs,
                                            int buf, int j0, int pk, int tid) {
  unsigned short* base = Bs + buf * PANEL_USH;
#pragma unroll
  for (int o = 0; o < 8; ++o) {
    const int L    = (tid + o * NTHR) * 8;   // ushort offset of this 16B chunk
    const int bb   = L >> 9;                 // frag block (kc*4 + g)
    const int wi   = L & 511;
    const int fl   = wi >> 4;                // frag lane 0..31
    const int half = (wi >> 3) & 1;
    const int g    = bb & 3;
    const int kc   = bb >> 2;
    const unsigned short* gp = wbf + (size_t)(g * H_ + j0 + (fl & 15)) * H_
                               + pk + kc * 32 + (fl >> 4) * 8 + half * 16;
    unsigned short* lp = base + L;
#if USE_ASYNC_LDS
    __builtin_amdgcn_global_load_async_to_lds_b128(
        (gbl_i32x4*)(void*)gp, (lds_i32x4*)(void*)lp, 0, 0);
#else
    *reinterpret_cast<v8bf*>(lp) = *reinterpret_cast<const v8bf*>(gp);
#endif
  }
}

// acc[0..3]: row tile r0,   gates 0..3
// acc[4..7]: row tile r0+16, gates 0..3
__device__ __forceinline__ void gemm_lds(const unsigned short* __restrict__ hbuf,
                                         const unsigned short* __restrict__ wbf,
                                         unsigned short* __restrict__ Bs,
                                         int r0, int j0, int tid, v8f (&acc)[8]) {
  const int lane = tid & 31;
  const int lm = lane & 15, lh = lane >> 4;
  const unsigned short* apA = hbuf + (size_t)(r0 + lm) * H_ + lh * 8;
  const unsigned short* apB = apA + (size_t)16 * H_;

  issue_panel(wbf, Bs, 0, j0, 0, tid);
  for (int p = 0; p < NP; ++p) {
    const int buf = p & 1;
    if (p + 1 < NP) {
      issue_panel(wbf, Bs, buf ^ 1, j0, (p + 1) * KP, tid);
      wait_async_panel();          // panel p landed (next panel still in flight)
    } else {
      wait_async_all();
    }
    __syncthreads();               // panel p visible to all waves
    const int pk = p * KP;
    __builtin_prefetch(apA + pk + KP, 0, 1);   // stream A rows
    __builtin_prefetch(apB + pk + KP, 0, 1);
#pragma unroll
    for (int kc = 0; kc < KP / 32; ++kc) {
      v16bf aA = load_afrag(apA + pk + kc * 32);
      v16bf aB = load_afrag(apB + pk + kc * 32);
      v16bf b0 = read_bfrag(Bs, buf, kc, 0, lane);
      v16bf b1 = read_bfrag(Bs, buf, kc, 1, lane);
      v16bf b2 = read_bfrag(Bs, buf, kc, 2, lane);
      v16bf b3 = read_bfrag(Bs, buf, kc, 3, lane);
      acc[0] = WMMA_BF16(aA, b0, acc[0]);
      acc[1] = WMMA_BF16(aA, b1, acc[1]);
      acc[2] = WMMA_BF16(aA, b2, acc[2]);
      acc[3] = WMMA_BF16(aA, b3, acc[3]);
      acc[4] = WMMA_BF16(aB, b0, acc[4]);
      acc[5] = WMMA_BF16(aB, b1, acc[5]);
      acc[6] = WMMA_BF16(aB, b2, acc[6]);
      acc[7] = WMMA_BF16(aB, b3, acc[7]);
    }
    __syncthreads();               // everyone done with buf before it is rewritten
  }
}

__global__ void zero_kernel(unsigned* counter, float* cz, int nc, unsigned short* hz, int nh) {
  int i0 = blockIdx.x * blockDim.x + threadIdx.x;
  int stride = gridDim.x * blockDim.x;
  if (i0 == 0) *counter = 0u;
  for (int i = i0; i < nc; i += stride) cz[i] = 0.0f;
  for (int i = i0; i < nh; i += stride) hz[i] = 0;
}

__global__ void cvt_kernel(const float* __restrict__ src, unsigned short* __restrict__ dst, int n) {
  int i0 = blockIdx.x * blockDim.x + threadIdx.x;
  int stride = gridDim.x * blockDim.x;
  for (int i = i0; i < n; i += stride) dst[i] = f2bf(src[i]);
}

__global__ __launch_bounds__(NTHR) void lstm_persistent_kernel(
    const float* __restrict__ X,     // [B, T]
    const float* __restrict__ Wih0,  // [4H, 1] fp32
    const float* __restrict__ bih0, const float* __restrict__ bhh0,
    const float* __restrict__ bih1, const float* __restrict__ bhh1,
    const float* __restrict__ Wlin,  // [H] fp32
    const float* __restrict__ blin,  // [1]
    const unsigned short* __restrict__ whh0b,  // bf16 [4H, H]
    const unsigned short* __restrict__ wih1b,  // bf16 [4H, H]
    const unsigned short* __restrict__ whh1b,  // bf16 [4H, H]
    unsigned short* __restrict__ h0buf,        // bf16 [2][B, H] ping-pong
    unsigned short* __restrict__ h1buf,        // bf16 [2][B, H] ping-pong
    float* __restrict__ c0g, float* __restrict__ c1g,   // fp32 [B, H]
    float* __restrict__ part,                  // [2][NBLK, B] partial dots
    unsigned* __restrict__ counter,
    float* __restrict__ out)                   // [B, T]
{
  __shared__ unsigned short Bs[2 * PANEL_USH]; // 32KB double-buffered B panels

  const int jt   = blockIdx.x;
  const int j0   = jt * 16;
  const int tid  = threadIdx.x;
  const int wv   = tid >> 5;
  const int lane = tid & 31;
  const int lm   = lane & 15;
  const int lh   = lane >> 4;
  const int r0   = wv * 32;           // this wave: rows [r0, r0+32)
  const int col  = j0 + lm;
  const float bl = *blin;

  // Per-lane gate-column constants (hoisted out of the time loop)
  const float b0i = bih0[col]        + bhh0[col];
  const float b0f = bih0[H_ + col]   + bhh0[H_ + col];
  const float b0g = bih0[2*H_ + col] + bhh0[2*H_ + col];
  const float b0o = bih0[3*H_ + col] + bhh0[3*H_ + col];
  const float w0i = Wih0[col],        w0f = Wih0[H_ + col];
  const float w0g = Wih0[2*H_ + col], w0o = Wih0[3*H_ + col];
  const float b1i = bih1[col]        + bhh1[col];
  const float b1f = bih1[H_ + col]   + bhh1[H_ + col];
  const float b1g = bih1[2*H_ + col] + bhh1[2*H_ + col];
  const float b1o = bih1[3*H_ + col] + bhh1[3*H_ + col];
  const float wl  = Wlin[col];

  unsigned phase = 0;

  for (int t = 0; t < T_; ++t) {
    const int rb = t & 1, wb = rb ^ 1;
    unsigned short* h0r = h0buf + (size_t)rb * B_ * H_;
    unsigned short* h0w = h0buf + (size_t)wb * B_ * H_;
    unsigned short* h1r = h1buf + (size_t)rb * B_ * H_;
    unsigned short* h1w = h1buf + (size_t)wb * B_ * H_;

    // ---- Layer 0: gates0 = h0 @ Whh0^T (+ rank-1 x term + biases) ----
    v8f g[8] = {};
    gemm_lds(h0r, whh0b, Bs, r0, j0, tid, g);
#pragma unroll
    for (int tl = 0; tl < 2; ++tl) {
#pragma unroll
      for (int v = 0; v < 8; ++v) {
        const int row = r0 + tl * 16 + lh * 8 + v;
        const float x = X[row * T_ + t];
        const float gi = sigm (g[tl*4+0][v] + b0i + x * w0i);
        const float gf = sigm (g[tl*4+1][v] + b0f + x * w0f);
        const float gg = tanh_(g[tl*4+2][v] + b0g + x * w0g);
        const float go = sigm (g[tl*4+3][v] + b0o + x * w0o);
        const size_t idx = (size_t)row * H_ + col;
        const float c = gf * c0g[idx] + gi * gg;
        c0g[idx] = c;
        h0w[idx] = f2bf(go * tanh_(c));
      }
    }
    grid_sync(counter, (unsigned)NBLK * (++phase));   // h0_new visible everywhere

    // ---- Layer 1: gates1 = h0_new @ Wih1^T + h1 @ Whh1^T (+ biases) ----
    v8f q[8] = {};
    gemm_lds(h0w, wih1b, Bs, r0, j0, tid, q);
    gemm_lds(h1r, whh1b, Bs, r0, j0, tid, q);
    float py[2][8];
#pragma unroll
    for (int tl = 0; tl < 2; ++tl) {
#pragma unroll
      for (int v = 0; v < 8; ++v) {
        const int row = r0 + tl * 16 + lh * 8 + v;
        const float gi = sigm (q[tl*4+0][v] + b1i);
        const float gf = sigm (q[tl*4+1][v] + b1f);
        const float gg = tanh_(q[tl*4+2][v] + b1g);
        const float go = sigm (q[tl*4+3][v] + b1o);
        const size_t idx = (size_t)row * H_ + col;
        const float c = gf * c1g[idx] + gi * gg;
        c1g[idx] = c;
        const float h = go * tanh_(c);
        h1w[idx] = f2bf(h);
        py[tl][v] = h * wl;   // partial of y[row] over this block's 16 columns
      }
    }
    // Deterministic cross-lane reduce over the 16 columns (lm dimension)
#pragma unroll
    for (int off = 1; off < 16; off <<= 1)
#pragma unroll
      for (int tl = 0; tl < 2; ++tl)
#pragma unroll
        for (int v = 0; v < 8; ++v) py[tl][v] += __shfl_xor(py[tl][v], off, 32);
    if (lm == 0) {
      float* pb = part + (size_t)rb * (NBLK * B_) + jt * B_;
#pragma unroll
      for (int tl = 0; tl < 2; ++tl)
#pragma unroll
        for (int v = 0; v < 8; ++v) pb[r0 + tl * 16 + lh * 8 + v] = py[tl][v];
    }
    grid_sync(counter, (unsigned)NBLK * (++phase));   // partials + h1_new visible

    // ---- Output head: block jt reduces batch rows {2jt, 2jt+1} ----
    if (tid < 2) {
      const int b = jt * 2 + tid;
      float s = bl;
      const float* pb = part + (size_t)rb * (NBLK * B_);
      for (int k = 0; k < NBLK; ++k) s += pb[k * B_ + b];
      out[(size_t)b * T_ + t] = s;
    }
  }
}

extern "C" void kernel_launch(void* const* d_in, const int* in_sizes, int n_in,
                              void* d_out, int out_size, void* d_ws, size_t ws_size,
                              hipStream_t stream) {
  const float* X    = (const float*)d_in[0];
  const float* Wih0 = (const float*)d_in[1];
  const float* Whh0 = (const float*)d_in[2];
  const float* bih0 = (const float*)d_in[3];
  const float* bhh0 = (const float*)d_in[4];
  const float* Wih1 = (const float*)d_in[5];
  const float* Whh1 = (const float*)d_in[6];
  const float* bih1 = (const float*)d_in[7];
  const float* bhh1 = (const float*)d_in[8];
  const float* Wlin = (const float*)d_in[9];
  const float* blin = (const float*)d_in[10];
  float* out = (float*)d_out;

  // Workspace layout (~27.3 MB)
  char* ws = (char*)d_ws;
  size_t off = 0;
  unsigned* counter = (unsigned*)(ws + off);           off += 256;
  const size_t WN = (size_t)4 * H_ * H_;               // 4096*1024 elements
  unsigned short* whh0b = (unsigned short*)(ws + off); off += WN * 2;
  unsigned short* wih1b = (unsigned short*)(ws + off); off += WN * 2;
  unsigned short* whh1b = (unsigned short*)(ws + off); off += WN * 2;
  unsigned short* h0buf = (unsigned short*)(ws + off); off += (size_t)2 * B_ * H_ * 2;
  unsigned short* h1buf = (unsigned short*)(ws + off); off += (size_t)2 * B_ * H_ * 2;
  float* c0g  = (float*)(ws + off);                    off += (size_t)B_ * H_ * 4;
  float* c1g  = (float*)(ws + off);                    off += (size_t)B_ * H_ * 4;
  float* part = (float*)(ws + off);                    off += (size_t)2 * NBLK * B_ * 4;

  // 1) zero counter, c-state (c0,c1 contiguous) and h ping-pong buffers (contiguous)
  zero_kernel<<<512, 256, 0, stream>>>(counter, c0g, 2 * B_ * H_, h0buf, 4 * B_ * H_);
  // 2) fp32 -> bf16 weight conversion (every launch; deterministic)
  cvt_kernel<<<4096, 256, 0, stream>>>(Whh0, whh0b, (int)WN);
  cvt_kernel<<<4096, 256, 0, stream>>>(Wih1, wih1b, (int)WN);
  cvt_kernel<<<4096, 256, 0, stream>>>(Whh1, whh1b, (int)WN);
  // 3) persistent recurrent kernel: 64 resident workgroups, device-wide barrier
  lstm_persistent_kernel<<<NBLK, NTHR, 0, stream>>>(
      X, Wih0, bih0, bhh0, bih1, bhh1, Wlin, blin,
      whh0b, wih1b, whh1b, h0buf, h1buf, c0g, c1g, part, counter, out);
}
